// VarianceAdaptor_10256381903532
// MI455X (gfx1250) — matrix operand order repaired and checked by reading, not compiled, not run on hardware
//
#include <hip/hip_runtime.h>
#include <hip/hip_bf16.h>

typedef __attribute__((ext_vector_type(16))) __bf16 v16bf;
typedef __attribute__((ext_vector_type(8)))  float  v8f;

#define BB 8
#define SS 2048
#define DD 512
#define HH 512
#define MAXLEN 8192

__device__ __forceinline__ unsigned short f2bf(float f) {
    union { float f; unsigned u; } v; v.f = f;
    unsigned u = v.u;
    unsigned r = (u + 0x7fffu + ((u >> 16) & 1u)) >> 16;
    return (unsigned short)r;
}

__device__ __forceinline__ float bf2f(unsigned short h) {
    union { unsigned u; float f; } v; v.u = ((unsigned)h) << 16;
    return v.f;
}

// ---------------------------------------------------------------------------
// Repack conv weight (3,512,512) f32 -> WMMA B-fragment order (bf16 pairs).
// pk[((((kk*16+db)*4+nb)*8+nsub)*32+lane)*8+v] = bf16(W[kk][d][n]) | bf16(W[kk][d+1][n])<<16
//   d = db*32 + (lane>>4)*16 + 2v,  n = nb*128 + nsub*16 + (lane&15)
// ---------------------------------------------------------------------------
__global__ void repack_w_kernel(const float* __restrict__ W, unsigned* __restrict__ pk) {
    int idx = blockIdx.x * 256 + threadIdx.x;          // 0 .. 393215
    if (idx >= 3 * 16 * 4 * 8 * 32 * 8) return;
    int v    = idx & 7;
    int lane = (idx >> 3) & 31;
    int nsub = (idx >> 8) & 7;
    int nb   = (idx >> 11) & 3;
    int db   = (idx >> 13) & 15;
    int kk   = idx >> 17;
    int d = db * 32 + ((lane >> 4) * 16) + 2 * v;
    int n = nb * 128 + nsub * 16 + (lane & 15);
    const float* base = W + ((size_t)(kk * DD + d) * HH + n);
    unsigned lo = f2bf(base[0]);
    unsigned hi = f2bf(base[HH]);                      // d+1 row
    pk[idx] = lo | (hi << 16);
}

// ---------------------------------------------------------------------------
// f32 -> bf16 pair conversion
// ---------------------------------------------------------------------------
__global__ void f32_to_bf16_kernel(const float* __restrict__ X, unsigned* __restrict__ Y, int n2) {
    int i = blockIdx.x * 256 + threadIdx.x;
    if (i >= n2) return;
    float a = X[2 * i], b = X[2 * i + 1];
    Y[i] = (unsigned)f2bf(a) | ((unsigned)f2bf(b) << 16);
}

// ---------------------------------------------------------------------------
// Conv1d-as-GEMM with WMMA bf16: Y[b,t,n] = relu(bias[n] + sum_{kk,d} A[b,t+kk-1,d]*W[kk,d,n])
// Block: 256 threads (8 waves), tile M=128 (t), N=128 (n). Waves in 2x4 grid,
// each wave owns a 64x32 region = 4(M) x 2(N) 16x16 subtiles.
// K-loop: 16 d-blocks of 32; per d-block one 130-row A tile (t0-1..t0+128)
// shared by all 3 conv taps (tap kk reads A rows offset by +kk), plus all
// 3 taps' B tiles staged together -> 24 WMMAs per wave per barrier stage.
// ---------------------------------------------------------------------------
__launch_bounds__(256)
__global__ void conv_gemm_kernel(const unsigned short* __restrict__ Abf,  // [B][S][512] bf16 bits
                                 const unsigned*       __restrict__ Wpk,  // packed fragments
                                 const float*          __restrict__ bias, // [512]
                                 float*                __restrict__ Y)    // [B][S][512] f32 (relu'd)
{
    __shared__ unsigned Alds[130 * 16];     // 130 rows x 32 bf16 (16 u32) = 8.3KB
    __shared__ unsigned Blds[3 * 2048];     // 3 taps x (32x128 bf16 fragment-ordered) = 24KB

    const int tid  = threadIdx.x;
    const int lane = tid & 31;
    const int w    = tid >> 5;
    const int wm   = w >> 2;        // 0..1
    const int wn   = w & 3;         // 0..3
    const int t0   = blockIdx.x * 128;
    const int nb   = blockIdx.y;
    const int b    = blockIdx.z;

    v8f acc[4][2];
#pragma unroll
    for (int i = 0; i < 4; ++i)
#pragma unroll
        for (int j = 0; j < 2; ++j) acc[i][j] = (v8f)0.f;

    const unsigned short* Ab = Abf + (size_t)b * SS * DD;

    for (int db = 0; db < 16; ++db) {
        // --- stage A tile: 130 rows (t0-1 .. t0+128) x 32 bf16, zero at edges
        {
            int h = tid;                        // half-row index 0..259
#pragma unroll
            for (int rep = 0; rep < 2; ++rep) {
                if (h < 260) {
                    int row  = h >> 1;
                    int half = h & 1;
                    int t    = t0 + row - 1;
                    uint4 z0 = make_uint4(0, 0, 0, 0), z1 = z0;
                    if ((unsigned)t < (unsigned)SS) {
                        const uint4* src = (const uint4*)(Ab + (size_t)t * DD + db * 32 + half * 16);
                        z0 = src[0]; z1 = src[1];
                    }
                    uint4* dst = (uint4*)&Alds[row * 16 + half * 8];
                    dst[0] = z0; dst[1] = z1;
                }
                h += 256;
            }
        }
        // --- stage B tiles for all 3 taps: 3 x 8KB contiguous copies
#pragma unroll
        for (int kk = 0; kk < 3; ++kk) {
            const uint4* src = (const uint4*)(Wpk + (((kk * 16 + db) * 4 + nb) << 11));
            uint4* dst = (uint4*)&Blds[kk * 2048];
            dst[tid]       = src[tid];
            dst[tid + 256] = src[tid + 256];
        }
        __syncthreads();

        const int hk = lane >> 4;
#pragma unroll
        for (int kk = 0; kk < 3; ++kk) {
            // B fragments for this wave's two N subtiles
            v16bf bfr[2];
#pragma unroll
            for (int ni = 0; ni < 2; ++ni) {
                int nsub = wn * 2 + ni;
                const uint4* bp = (const uint4*)&Blds[kk * 2048 + (nsub * 32 + lane) * 8];
                union { uint4 q[2]; v16bf v; } ub;
                ub.q[0] = bp[0]; ub.q[1] = bp[1];
                bfr[ni] = ub.v;
            }
            // 4 M subtiles, A rows shifted by +kk (LDS row 0 == t0-1)
#pragma unroll
            for (int mi = 0; mi < 4; ++mi) {
                int arow = wm * 64 + mi * 16 + (lane & 15) + kk;
                const uint4* ap = (const uint4*)&Alds[arow * 16];
                union { uint4 q[2]; v16bf v; } ua;
                ua.q[0] = ap[2 * hk];
                ua.q[1] = ap[2 * hk + 1];
                v16bf afr = ua.v;
                acc[mi][0] = __builtin_amdgcn_wmma_f32_16x16x32_bf16(
                    false, afr, false, bfr[0], (short)0, acc[mi][0], false, false);
                acc[mi][1] = __builtin_amdgcn_wmma_f32_16x16x32_bf16(
                    false, afr, false, bfr[1], (short)0, acc[mi][1], false, false);
            }
        }
        __syncthreads();
    }

    // --- epilogue: bias + relu, C layout lane=(N, M-half), vgpr v -> M = (lane>>4)*8 + v
#pragma unroll
    for (int mi = 0; mi < 4; ++mi) {
#pragma unroll
        for (int ni = 0; ni < 2; ++ni) {
            int nsub  = wn * 2 + ni;
            int n     = nb * 128 + nsub * 16 + (lane & 15);
            float bv  = bias[n];
            int mbase = wm * 64 + mi * 16 + (lane >> 4) * 8;
            float* yp = Y + ((size_t)b * SS + t0 + mbase) * HH + n;
            v8f a = acc[mi][ni];
#pragma unroll
            for (int v = 0; v < 8; ++v) {
                float val = a[v] + bv;
                val = val > 0.f ? val : 0.f;
                yp[(size_t)v * HH] = val;
            }
        }
    }
}

// ---------------------------------------------------------------------------
// LayerNorm over last dim (512), writes bf16 (for next WMMA GEMM / proj)
// ---------------------------------------------------------------------------
__launch_bounds__(256)
__global__ void ln_kernel(const float* __restrict__ X, const float* __restrict__ gw,
                          const float* __restrict__ gb, unsigned short* __restrict__ Ybf)
{
    int row = blockIdx.x;
    const float* xr = X + (size_t)row * HH;
    int tid = threadIdx.x;
    float x0 = xr[tid], x1 = xr[tid + 256];
    float s  = x0 + x1;
    float s2 = x0 * x0 + x1 * x1;
#pragma unroll
    for (int off = 16; off; off >>= 1) {
        s  += __shfl_down(s,  off, 32);
        s2 += __shfl_down(s2, off, 32);
    }
    __shared__ float sh[16];
    int lane = tid & 31, wv = tid >> 5;
    if (lane == 0) { sh[wv] = s; sh[wv + 8] = s2; }
    __syncthreads();
    if (tid == 0) {
        float ts = 0.f, ts2 = 0.f;
#pragma unroll
        for (int i = 0; i < 8; ++i) { ts += sh[i]; ts2 += sh[i + 8]; }
        sh[0] = ts; sh[1] = ts2;
    }
    __syncthreads();
    float mu  = sh[0] * (1.f / HH);
    float var = sh[1] * (1.f / HH) - mu * mu;
    float rs  = rsqrtf(var + 1e-5f);
    float y0 = (x0 - mu) * rs * gw[tid]       + gb[tid];
    float y1 = (x1 - mu) * rs * gw[tid + 256] + gb[tid + 256];
    Ybf[(size_t)row * HH + tid]       = f2bf(y0);
    Ybf[(size_t)row * HH + tid + 256] = f2bf(y1);
}

// ---------------------------------------------------------------------------
// Projection to 1: out[row] = dot(h[row], proj_w) + proj_b ; one wave per row
// ---------------------------------------------------------------------------
__launch_bounds__(256)
__global__ void proj_kernel(const unsigned short* __restrict__ Hbf,
                            const float* __restrict__ pw, const float* __restrict__ pb,
                            float* __restrict__ out)
{
    int row  = blockIdx.x * 8 + (threadIdx.x >> 5);
    int lane = threadIdx.x & 31;
    const unsigned short* hr = Hbf + (size_t)row * HH;
    float s = 0.f;
#pragma unroll
    for (int i = lane; i < HH; i += 32) s += bf2f(hr[i]) * pw[i];
#pragma unroll
    for (int off = 16; off; off >>= 1) s += __shfl_down(s, off, 32);
    if (lane == 0) out[row] = s + pb[0];
}

// ---------------------------------------------------------------------------
// bucketize (searchsorted left over 255 bins) + embedding add.
// Xout = Xin + emb[idx]; optional bf16 mirror for next predictor input.
// ---------------------------------------------------------------------------
__launch_bounds__(256)
__global__ void bucket_add_kernel(const float* __restrict__ Xin, const float* __restrict__ vals,
                                  const float* __restrict__ bins, const float* __restrict__ emb,
                                  float* __restrict__ Xout, unsigned short* __restrict__ Xbf)
{
    int row = blockIdx.x;
    __shared__ int sidx;
    if (threadIdx.x == 0) {
        float v = vals[row];
        int idx = 0;
        for (int i = 0; i < 255; ++i) idx += (bins[i] < v) ? 1 : 0;
        sidx = idx;
    }
    __syncthreads();
    int idx = sidx;
    const float* er = emb + (size_t)idx * DD;
    const float* xr = Xin + (size_t)row * DD;
    float* yr = Xout + (size_t)row * DD;
    for (int i = threadIdx.x; i < DD; i += 256) {
        float o = xr[i] + er[i];
        yr[i] = o;
        if (Xbf) Xbf[(size_t)row * DD + i] = f2bf(o);
    }
}

// ---------------------------------------------------------------------------
// Per-batch cumsum of durations (S=2048) + out_lens (stored as float value)
// ---------------------------------------------------------------------------
__global__ void cumsum_kernel(const int* __restrict__ dur, int* __restrict__ cum,
                              float* __restrict__ lens_out)
{
    int b = threadIdx.x;
    if (b >= BB) return;
    const int* dr = dur + b * SS;
    int* cr = cum + b * SS;
    int s = 0;
    for (int i = 0; i < SS; ++i) { s += dr[i]; cr[i] = s; }
    lens_out[b] = (float)s;
}

// ---------------------------------------------------------------------------
// Length regulate: for each (b,pos) binary-search cum, gather row or zero.
// One 128-thread block per output row (128 x float4 = 512 floats).
// ---------------------------------------------------------------------------
__launch_bounds__(128)
__global__ void length_reg_kernel(const float* __restrict__ Xf, const int* __restrict__ cum,
                                  float* __restrict__ Out)
{
    int b   = blockIdx.y;
    int pos = blockIdx.x;
    const int* cr = cum + b * SS;
    int total = cr[SS - 1];
    int lo = 0, hi = SS;                 // searchsorted right
    while (lo < hi) { int mid = (lo + hi) >> 1; if (cr[mid] <= pos) lo = mid + 1; else hi = mid; }
    int idx = lo < SS - 1 ? lo : SS - 1;
    bool valid = pos < total;
    const float4* src = (const float4*)(Xf + ((size_t)b * SS + idx) * DD);
    float4* dst = (float4*)(Out + ((size_t)b * MAXLEN + pos) * DD);
    float4 z = make_float4(0.f, 0.f, 0.f, 0.f);
    dst[threadIdx.x] = valid ? src[threadIdx.x] : z;
}

// ---------------------------------------------------------------------------
// host launcher
// ---------------------------------------------------------------------------
extern "C" void kernel_launch(void* const* d_in, const int* in_sizes, int n_in,
                              void* d_out, int out_size, void* d_ws, size_t ws_size,
                              hipStream_t stream) {
    (void)in_sizes; (void)n_in; (void)out_size; (void)ws_size;

    // input order: x, padding_mask, durations, max_len, then 3 predictor param
    // dicts (10 leaves each), embed_pitch, embed_energy, pitch_bins, energy_bins
    const float* x          = (const float*)d_in[0];
    const int*   durations  = (const int*)d_in[2];
    const int P0 = 4;   // dur_params
    const int P1 = 14;  // pitch_params
    const int P2 = 24;  // energy_params
    const float* embed_pitch  = (const float*)d_in[34];
    const float* embed_energy = (const float*)d_in[35];
    const float* pitch_bins   = (const float*)d_in[36];
    const float* energy_bins  = (const float*)d_in[37];

    float* out = (float*)d_out;
    const size_t OFF_LENS   = (size_t)BB * MAXLEN * DD;   // 33554432
    const size_t OFF_LOGDUR = OFF_LENS + BB;
    const size_t OFF_PITCH  = OFF_LOGDUR + (size_t)BB * SS;
    const size_t OFF_ENERGY = OFF_PITCH + (size_t)BB * SS;

    // workspace carve-up
    char* ws = (char*)d_ws;
    size_t off = 0;
    auto alloc = [&](size_t bytes) -> void* {
        void* p = ws + off;
        off = (off + bytes + 255) & ~(size_t)255;
        return p;
    };
    const size_t PK_U32 = 3 * 16 * 4 * 8 * 32 * 8;        // 393216 u32 per tensor
    unsigned* wpk[6];
    for (int i = 0; i < 6; ++i) wpk[i] = (unsigned*)alloc(PK_U32 * 4);
    unsigned short* xa_bf = (unsigned short*)alloc((size_t)BB * SS * DD * 2); // bf16 predictor input
    float*          h_f32 = (float*)alloc((size_t)BB * SS * HH * 4);          // GEMM output ping
    unsigned short* h_bf  = (unsigned short*)alloc((size_t)BB * SS * HH * 2); // LN output pong
    float*          x2    = (float*)alloc((size_t)BB * SS * DD * 4);          // x + embeddings (f32)
    int*            cum   = (int*)alloc((size_t)BB * SS * 4);

    // 1) repack conv weights into WMMA B-fragment layout
    const float* convw[6] = {
        (const float*)d_in[P0 + 0], (const float*)d_in[P0 + 4],
        (const float*)d_in[P1 + 0], (const float*)d_in[P1 + 4],
        (const float*)d_in[P2 + 0], (const float*)d_in[P2 + 4],
    };
    for (int i = 0; i < 6; ++i)
        repack_w_kernel<<<(PK_U32 + 255) / 256, 256, 0, stream>>>(convw[i], wpk[i]);

    // 2) x -> bf16
    {
        int n2 = BB * SS * DD / 2;
        f32_to_bf16_kernel<<<(n2 + 255) / 256, 256, 0, stream>>>(x, (unsigned*)xa_bf, n2);
    }

    auto run_predictor = [&](const unsigned short* in_bf, unsigned* w1, unsigned* w2,
                             int P, float* out_rows) {
        const float* c1b = (const float*)d_in[P + 1];
        const float* l1w = (const float*)d_in[P + 2];
        const float* l1b = (const float*)d_in[P + 3];
        const float* c2b = (const float*)d_in[P + 5];
        const float* l2w = (const float*)d_in[P + 6];
        const float* l2b = (const float*)d_in[P + 7];
        const float* pw  = (const float*)d_in[P + 8];
        const float* pb  = (const float*)d_in[P + 9];
        conv_gemm_kernel<<<dim3(SS / 128, HH / 128, BB), 256, 0, stream>>>(in_bf, w1, c1b, h_f32);
        ln_kernel<<<BB * SS, 256, 0, stream>>>(h_f32, l1w, l1b, h_bf);
        conv_gemm_kernel<<<dim3(SS / 128, HH / 128, BB), 256, 0, stream>>>(h_bf, w2, c2b, h_f32);
        ln_kernel<<<BB * SS, 256, 0, stream>>>(h_f32, l2w, l2b, h_bf);
        proj_kernel<<<BB * SS / 8, 256, 0, stream>>>(h_bf, pw, pb, out_rows);
    };

    // 3) duration + pitch predictors on original x
    run_predictor(xa_bf, wpk[0], wpk[1], P0, out + OFF_LOGDUR);
    run_predictor(xa_bf, wpk[2], wpk[3], P1, out + OFF_PITCH);

    // 4) x2 = x + embed_pitch[bucket(pitch_out)]; bf16 mirror reuses xa_bf
    bucket_add_kernel<<<BB * SS, 256, 0, stream>>>(x, out + OFF_PITCH, pitch_bins,
                                                   embed_pitch, x2, xa_bf);

    // 5) energy predictor on updated x
    run_predictor(xa_bf, wpk[4], wpk[5], P2, out + OFF_ENERGY);

    // 6) x2 += embed_energy[bucket(energy_out)] (in place, no bf16 needed)
    bucket_add_kernel<<<BB * SS, 256, 0, stream>>>(x2, out + OFF_ENERGY, energy_bins,
                                                   embed_energy, x2, nullptr);

    // 7) cumsum + out_lens
    cumsum_kernel<<<1, 32, 0, stream>>>(durations, cum, out + OFF_LENS);

    // 8) length regulate -> x_out
    length_reg_kernel<<<dim3(MAXLEN, BB), 128, 0, stream>>>(x2, cum, out);
}
